// ATT_13657996001412
// MI455X (gfx1250) — matrix operand order repaired and checked by reading, not compiled, not run on hardware
//
#include <hip/hip_runtime.h>
#include <hip/hip_bf16.h>

// ---------------------------------------------------------------------------
// Problem constants (from reference): B=4, S=4096, E=256, H=4, dh=64
// ---------------------------------------------------------------------------
constexpr int BATCH = 4;
constexpr int SEQ   = 4096;
constexpr int EMB   = 256;
constexpr int HEADS = 4;
constexpr int DH    = 64;          // head dim
constexpr int MROWS = BATCH * SEQ; // 16384

typedef __attribute__((ext_vector_type(16))) __bf16 v16bf;
typedef __attribute__((ext_vector_type(8)))  __bf16 v8bf;
typedef __attribute__((ext_vector_type(4)))  __bf16 v4bf;
typedef __attribute__((ext_vector_type(8)))  float  v8f;

#define WMMA_BF16(A, B, C) \
  __builtin_amdgcn_wmma_f32_16x16x32_bf16(false, (A), false, (B), (short)0, (C), false, false)

// ---------------------------------------------------------------------------
// Tensor Data Mover: build a D# (CDNA5 ISA) and issue
// tensor_load_to_lds. Used for tiles that need no transpose/convert.
//   - group0: count=1 | lds_addr | global_addr | type=2
//   - group1: data_size=1 (2B elems), tensor/tile dims, dim0 stride
// Builtin arity differs by toolchain (probe-verified):
//   clang-22 (ROCm 7.2): 5 args; clang-23 (amdgpu-toolchain): 6 args.
// ---------------------------------------------------------------------------
#if defined(__has_builtin)
#if __has_builtin(__builtin_amdgcn_tensor_load_to_lds)
#define HAVE_TDM 1
#endif
#endif

#ifdef HAVE_TDM
typedef __attribute__((ext_vector_type(4))) unsigned int v4u;
typedef __attribute__((ext_vector_type(8))) int          v8i_t;
typedef __attribute__((ext_vector_type(4))) int          v4i_t;

// 2-D tile load (elements are 2 bytes). tile_d1==0 -> 1-D tile.
__device__ __forceinline__ void tdm_load_2d(const void* gsrc, unsigned lds_off,
                                            unsigned tile_d0, unsigned tile_d1,
                                            unsigned tensor_d0, unsigned tensor_d1,
                                            unsigned stride0) {
  const unsigned long long ga = (unsigned long long)(size_t)gsrc;
  v4u g0;
  g0[0] = 1u;                                              // count=1 (valid D#)
  g0[1] = lds_off;                                         // LDS byte address
  g0[2] = (unsigned)ga;                                    // global_addr[31:0]
  g0[3] = (unsigned)((ga >> 32) & 0x1ffffffu) | (2u << 30); // addr[56:32] | type=2
  v8i_t g1;
  g1[0] = (int)(1u << 16);                                 // data_size=1 -> 2 bytes
  g1[1] = (int)((tensor_d0 & 0xffffu) << 16);              // tensor_dim0[15:0]
  g1[2] = (int)((tensor_d0 >> 16) | ((tensor_d1 & 0xffffu) << 16));
  g1[3] = (int)((tensor_d1 >> 16) | (tile_d0 << 16));      // tile_dim0
  g1[4] = (int)tile_d1;                                    // tile_dim1 (0=1-D)
  g1[5] = (int)stride0;                                    // tensor_dim0_stride lo
  g1[6] = 0;
  g1[7] = 0;
  const v4i_t gz = {0, 0, 0, 0};
#if __clang_major__ >= 23
  const v8i_t gz8 = {0, 0, 0, 0, 0, 0, 0, 0};
  __builtin_amdgcn_tensor_load_to_lds(g0, g1, gz, gz, gz8, 0);
#else
  __builtin_amdgcn_tensor_load_to_lds(g0, g1, gz, gz, 0);
#endif
}
__device__ __forceinline__ unsigned lds_offset_of(const void* p) {
  return (unsigned)(size_t)p;  // flat shared addr low 32 bits == LDS offset
}
#endif // HAVE_TDM

// ---------------------------------------------------------------------------
// DPP16 butterfly reductions across the 16-lane row (no LDS traffic).
// quad_perm[1,0,3,2]=0xB1 (xor1), quad_perm[2,3,0,1]=0x4E (xor2),
// row_half_mirror=0x141 (xor within 8), row_mirror=0x140 (xor within 16).
// ---------------------------------------------------------------------------
template <int CTRL>
__device__ __forceinline__ float dpp_mov_f32(float x) {
  return __int_as_float(
      __builtin_amdgcn_update_dpp(0, __float_as_int(x), CTRL, 0xf, 0xf, true));
}
__device__ __forceinline__ float row16_max(float v) {
  v = fmaxf(v, dpp_mov_f32<0x0B1>(v));
  v = fmaxf(v, dpp_mov_f32<0x04E>(v));
  v = fmaxf(v, dpp_mov_f32<0x141>(v));
  v = fmaxf(v, dpp_mov_f32<0x140>(v));
  return v;
}
__device__ __forceinline__ float row16_sum(float v) {
  v += dpp_mov_f32<0x0B1>(v);
  v += dpp_mov_f32<0x04E>(v);
  v += dpp_mov_f32<0x141>(v);
  v += dpp_mov_f32<0x140>(v);
  return v;
}

// ---------------------------------------------------------------------------
// WMMA fragment loaders (wave32 layouts, CDNA5 ISA §7.12.2).
// ---------------------------------------------------------------------------
__device__ __forceinline__ v16bf ld_a_frag(const __bf16* row32, int half) {
  v8bf lo = *(const v8bf*)(row32 + 8 * half);
  v8bf hi = *(const v8bf*)(row32 + 16 + 8 * half);
  v16bf a;
#pragma unroll
  for (int e = 0; e < 8; ++e) { a[e] = lo[e]; a[e + 8] = hi[e]; }
  return a;
}
__device__ __forceinline__ v16bf ld_b_frag(const __bf16* colrow32, int khalf) {
  v8bf lo = *(const v8bf*)(colrow32 + 8 * khalf);
  v8bf hi = *(const v8bf*)(colrow32 + 16 + 8 * khalf);
  v16bf b;
#pragma unroll
  for (int e = 0; e < 8; ++e) { b[e] = lo[e]; b[e + 8] = hi[e]; }
  return b;
}

// ---------------------------------------------------------------------------
// GEMM: Y(M x 256) = X(M x 256) @ W(256 x 256)
//   XF32:     X fp32 (converted while staging); else X bf16 (staged via TDM).
//   SPLITOUT: Y bf16 head-split [B,H,S,dh]; else fp32 flat [M,256].
// 128 threads (4 waves); block tile 64x64; K-step 32; 4 WMMA/wave/step.
// ---------------------------------------------------------------------------
template <bool XF32, bool SPLITOUT>
__global__ __launch_bounds__(128)
void gemm256_kernel(const void* __restrict__ Xv, const float* __restrict__ W,
                    void* __restrict__ Yv) {
  __shared__ alignas(16) __bf16 Xs[64][32];   // M-major
  __shared__ alignas(16) __bf16 Wt[64][32];   // transposed: Wt[n][k]

  const int tid  = threadIdx.x;
  const int wv   = tid >> 5;
  const int lane = tid & 31;
  const int half = lane >> 4;
  const int ln   = lane & 15;
  const int m0   = blockIdx.x * 64;
  const int n0   = blockIdx.y * 64;

  v8f acc[4] = {};

  for (int ks = 0; ks < EMB / 32; ++ks) {
    const int k0 = ks * 32;

    // ---- stage X tile (64x32) ----
    if (XF32) {
      const float* X = (const float*)Xv;
      const int r  = tid >> 3;          // 0..15
      const int c4 = (tid & 7) << 2;    // 0..28
#pragma unroll
      for (int t = 0; t < 4; ++t) {
        const int row = r + 16 * t;
        const float4 v = *(const float4*)(X + (size_t)(m0 + row) * EMB + k0 + c4);
        Xs[row][c4 + 0] = (__bf16)v.x; Xs[row][c4 + 1] = (__bf16)v.y;
        Xs[row][c4 + 2] = (__bf16)v.z; Xs[row][c4 + 3] = (__bf16)v.w;
      }
    } else {
#ifdef HAVE_TDM
      if (wv == 0) {
        const __bf16* X = (const __bf16*)Xv;
        // 2-D tile: 32 elems x 64 rows from a row-major [M][256] tensor.
        tdm_load_2d(X + (size_t)m0 * EMB + k0, lds_offset_of(&Xs[0][0]),
                    /*tile_d0=*/32, /*tile_d1=*/64,
                    /*tensor_d0=*/EMB, /*tensor_d1=*/64, /*stride0=*/EMB);
      }
#else
      const __bf16* X = (const __bf16*)Xv;
      const int r  = tid >> 3;
      const int c4 = (tid & 7) << 2;
#pragma unroll
      for (int t = 0; t < 4; ++t) {
        const int row = r + 16 * t;
        v4bf v = *(const v4bf*)(X + (size_t)(m0 + row) * EMB + k0 + c4);
        *(v4bf*)&Xs[row][c4] = v;
      }
#endif
    }
    // ---- stage W tile transposed (32x64 -> Wt[64][32]) ----
    {
      const int kk = tid & 31;          // 0..31
      const int ng = tid >> 5;          // 0..3
#pragma unroll
      for (int t = 0; t < 4; ++t) {
        const int nb = ng * 4 + 16 * t; // 0..60, step 4
        const float4 v = *(const float4*)(W + (size_t)(k0 + kk) * EMB + n0 + nb);
        Wt[nb + 0][kk] = (__bf16)v.x; Wt[nb + 1][kk] = (__bf16)v.y;
        Wt[nb + 2][kk] = (__bf16)v.z; Wt[nb + 3][kk] = (__bf16)v.w;
      }
    }
#ifdef HAVE_TDM
    if (!XF32 && wv == 0) __builtin_amdgcn_s_wait_tensorcnt(0);
#endif
    __syncthreads();

    const v16bf a = ld_a_frag(&Xs[wv * 16 + ln][0], half);
#pragma unroll
    for (int j = 0; j < 4; ++j) {
      const v16bf b = ld_b_frag(&Wt[16 * j + ln][0], half);
      acc[j] = WMMA_BF16(a, b, acc[j]);
    }
    __syncthreads();
  }

  // ---- store ----
#pragma unroll
  for (int j = 0; j < 4; ++j) {
#pragma unroll
    for (int r = 0; r < 8; ++r) {
      const int grow = m0 + wv * 16 + r + 8 * half;
      const int col  = n0 + 16 * j + ln;
      if (SPLITOUT) {
        const int b  = grow >> 12;        // / SEQ
        const int s  = grow & (SEQ - 1);
        const int h  = col >> 6;          // / DH
        const int d  = col & (DH - 1);
        ((__bf16*)Yv)[(((size_t)b * HEADS + h) * SEQ + s) * DH + d] = (__bf16)acc[j][r];
      } else {
        ((float*)Yv)[(size_t)grow * EMB + col] = acc[j][r];
      }
    }
  }
}

// ---------------------------------------------------------------------------
// Flash attention per (b, h): online-softmax over kv tiles of 64.
// 128 threads (4 waves); q-tile 64 rows; wave owns 16 q-rows.
// K tiles (8 KB, contiguous) arrive via TDM, double-buffered so the DMA for
// tile kt+1 overlaps the whole compute of tile kt. V is staged transposed
// manually (TDM cannot transpose). Scale 1/sqrt(dh)=0.125 folded into Q.
// ---------------------------------------------------------------------------
__global__ __launch_bounds__(128)
void attn_kernel(const __bf16* __restrict__ Q, const __bf16* __restrict__ K,
                 const __bf16* __restrict__ V, __bf16* __restrict__ O) {
  __shared__ alignas(16) __bf16 Ks[2][64][64];   // double-buffered K tile [kv][d]
  __shared__ alignas(16) __bf16 Vt[64][64];      // V tile transposed [d][kv]
  __shared__ alignas(16) __bf16 Pl[4][16][64];   // per-wave P (D->A relayout)

  const int tid  = threadIdx.x;
  const int wv   = tid >> 5;
  const int lane = tid & 31;
  const int half = lane >> 4;
  const int ln   = lane & 15;

  const int q0 = blockIdx.x * 64;
  const int bh = blockIdx.y;            // b*HEADS + h
  const int b  = bh >> 2;
  const int h  = bh & (HEADS - 1);
  const size_t head = (size_t)bh * SEQ * DH;
  const __bf16* Khead = K + head;
  const __bf16* Vhead = V + head;

#ifdef HAVE_TDM
  // Prologue: DMA first K tile (contiguous 4096 bf16 = one 1-D descriptor).
  if (wv == 0)
    tdm_load_2d(Khead, lds_offset_of(&Ks[0][0][0]),
                /*tile_d0=*/4096, /*tile_d1=*/0,
                /*tensor_d0=*/4096, /*tensor_d1=*/0, /*stride0=*/4096);
#endif

  // Q fragments (global A-layout loads; rows contiguous in dh), pre-scaled.
  v16bf qa[2];
  {
    const __bf16* qrow = Q + head + (size_t)(q0 + wv * 16 + ln) * DH;
#pragma unroll
    for (int kk = 0; kk < 2; ++kk) {
      v16bf t = ld_a_frag(qrow + 32 * kk, half);
#pragma unroll
      for (int e = 0; e < 16; ++e) t[e] = (__bf16)((float)t[e] * 0.125f);
      qa[kk] = t;
    }
  }

  float mrun[8], lrun[8];
  v8f oacc[4] = {};
#pragma unroll
  for (int r = 0; r < 8; ++r) { mrun[r] = -1e30f; lrun[r] = 0.0f; }

  for (int kt = 0; kt < SEQ / 64; ++kt) {
    const int kv0 = kt * 64;
    const int buf = kt & 1;

    // ---- stage V transposed (manual) ----
#pragma unroll
    for (int t = 0; t < 8; ++t) {
      const int cid = tid + 128 * t;      // 1024 chunks of 4 elems
      const int r   = cid >> 4;           // 0..63
      const int c4  = (cid & 15) << 2;    // 0..60
      v4bf vv = *(const v4bf*)(Vhead + (size_t)(kv0 + r) * DH + c4);
      Vt[c4 + 0][r] = vv[0]; Vt[c4 + 1][r] = vv[1];
      Vt[c4 + 2][r] = vv[2]; Vt[c4 + 3][r] = vv[3];
#ifndef HAVE_TDM
      v4bf kk2 = *(const v4bf*)(Khead + (size_t)(kv0 + r) * DH + c4);
      *(v4bf*)&Ks[buf][r][c4] = kk2;
#endif
    }
    // prefetch next V tile into cache (global_prefetch_b8)
    if (kt + 1 < SEQ / 64)
      __builtin_prefetch(Vhead + (size_t)(kv0 + 64 + (tid & 63)) * DH, 0, 0);

#ifdef HAVE_TDM
    if (wv == 0) __builtin_amdgcn_s_wait_tensorcnt(0);  // Ks[buf] ready
#endif
    __syncthreads();
#ifdef HAVE_TDM
    // Overlap: DMA next K tile into the other buffer while computing this one.
    if (wv == 0 && kt + 1 < SEQ / 64)
      tdm_load_2d(Khead + (size_t)(kv0 + 64) * DH,
                  lds_offset_of(&Ks[buf ^ 1][0][0]),
                  4096, 0, 4096, 0, 4096);
#endif

    // ---- scores S = (Q*scale) @ K^T : 16 x 64 per wave ----
    v8f sc[4];
#pragma unroll
    for (int j = 0; j < 4; ++j) {
      v8f z = {};
#pragma unroll
      for (int kk = 0; kk < 2; ++kk) {
        const v16bf kb = ld_b_frag(&Ks[buf][16 * j + ln][32 * kk], half);
        z = WMMA_BF16(qa[kk], kb, z);
      }
      sc[j] = z;
    }

    // ---- online softmax (rows in 16-lane halves; DPP butterfly reduce) ----
#pragma unroll
    for (int r = 0; r < 8; ++r) {
      float mx = fmaxf(fmaxf(sc[0][r], sc[1][r]), fmaxf(sc[2][r], sc[3][r]));
      mx = row16_max(mx);
      const float mn    = fmaxf(mrun[r], mx);
      const float alpha = __expf(mrun[r] - mn);
      mrun[r] = mn;
      float rs = 0.0f;
#pragma unroll
      for (int j = 0; j < 4; ++j) {
        const float p = __expf(sc[j][r] - mn);
        sc[j][r] = p;
        rs += p;
      }
      rs = row16_sum(rs);
      lrun[r] = lrun[r] * alpha + rs;
#pragma unroll
      for (int j = 0; j < 4; ++j) {
        oacc[j][r] *= alpha;
        Pl[wv][r + 8 * half][16 * j + ln] = (__bf16)sc[j][r];
      }
    }
    __syncthreads();   // P visible for A-layout reload

    // ---- O += P @ V ----
#pragma unroll
    for (int kk = 0; kk < 2; ++kk) {
      const v16bf pa = ld_a_frag(&Pl[wv][ln][32 * kk], half);
#pragma unroll
      for (int j = 0; j < 4; ++j) {
        const v16bf vb = ld_b_frag(&Vt[16 * j + ln][32 * kk], half);
        oacc[j] = WMMA_BF16(pa, vb, oacc[j]);
      }
    }
    __syncthreads();   // protect Vt/Pl (and Ks buffer rotation) for next iter
  }

  // ---- normalize and store O (merge heads: [B,S,E]) ----
#pragma unroll
  for (int j = 0; j < 4; ++j) {
#pragma unroll
    for (int r = 0; r < 8; ++r) {
      const int qrow = q0 + wv * 16 + r + 8 * half;
      const int col  = h * DH + 16 * j + ln;
      O[((size_t)b * SEQ + qrow) * EMB + col] = (__bf16)(oacc[j][r] / lrun[r]);
    }
  }
}

// ---------------------------------------------------------------------------
// Launch: 3 projections -> attention -> output projection, all on `stream`.
// Workspace: Q,K,V (bf16 [B,H,S,dh]) + O (bf16 [B,S,E]) = 32 MiB.
// ---------------------------------------------------------------------------
extern "C" void kernel_launch(void* const* d_in, const int* in_sizes, int n_in,
                              void* d_out, int out_size, void* d_ws, size_t ws_size,
                              hipStream_t stream) {
  (void)in_sizes; (void)n_in; (void)out_size; (void)ws_size;
  const float* target = (const float*)d_in[0];
  const float* source = (const float*)d_in[1];
  const float* Wq     = (const float*)d_in[2];
  const float* Wk     = (const float*)d_in[3];
  const float* Wv     = (const float*)d_in[4];
  const float* Wo     = (const float*)d_in[5];

  __bf16* Qw = (__bf16*)d_ws;
  __bf16* Kw = Qw + (size_t)MROWS * EMB;
  __bf16* Vw = Kw + (size_t)MROWS * EMB;
  __bf16* Ow = Vw + (size_t)MROWS * EMB;

  dim3 blk(128);
  dim3 gg(MROWS / 64, EMB / 64);    // 256 x 4 blocks
  dim3 ga(SEQ / 64, BATCH * HEADS); // 64 x 16 blocks

  gemm256_kernel<true,  true ><<<gg, blk, 0, stream>>>(target, Wq, Qw);
  gemm256_kernel<true,  true ><<<gg, blk, 0, stream>>>(source, Wk, Kw);
  gemm256_kernel<true,  true ><<<gg, blk, 0, stream>>>(source, Wv, Vw);
  attn_kernel<<<ga, blk, 0, stream>>>(Qw, Kw, Vw, Ow);
  gemm256_kernel<false, false><<<gg, blk, 0, stream>>>(Ow, Wo, (float*)d_out);
}
